// WindowedCausalSelfAttention_6588479832156
// MI455X (gfx1250) — matrix operand order repaired and checked by reading, not compiled, not run on hardware
//
#include <hip/hip_runtime.h>

#define B_ 4
#define T_ 2048
#define C_ 1024
#define H_ 16
#define D_ 64
#define WIN_ 256
#define M_ (B_*T_)      // 8192 rows (B*T)
#define N3_ (3*C_)      // 3072 qkv outputs

typedef _Float16 f16;
typedef __attribute__((ext_vector_type(16))) _Float16 v16h;
typedef __attribute__((ext_vector_type(8)))  _Float16 v8h;
typedef __attribute__((ext_vector_type(8)))  float    v8f;
typedef __attribute__((ext_vector_type(4)))  float    v4f;

static __device__ __forceinline__ v8f zero8() {
  v8f z;
#pragma unroll
  for (int i = 0; i < 8; ++i) z[i] = 0.f;
  return z;
}

static __device__ __forceinline__ v8f wmma16(v16h a, v16h b, v8f c) {
  // D = A(16x32 f16) * B(32x16 f16) + C(16x16 f32)
  return __builtin_amdgcn_wmma_f32_16x16x32_f16(false, a, false, b, (short)0, c,
                                                false, false);
}

// ---- A fragment (16x32, f16). Lane holds row (lane&15).
// halves 0..7  = K[kb8 .. kb8+7],  halves 8..15 = K[16+kb8 .. 16+kb8+7]
static __device__ __forceinline__ v16h afrag_f16(const f16* p, int kb8) {
  v8h a = *(const v8h*)(p + kb8);
  v8h b = *(const v8h*)(p + 16 + kb8);
  v16h r;
#pragma unroll
  for (int i = 0; i < 8; ++i) { r[i] = a[i]; r[i + 8] = b[i]; }
  return r;
}

// ---- B fragment (32x16, f16). Lane holds column (lane&15).
// halves 0..7 = K[hi16/2 .. hi16/2+7], halves 8..15 = K[16+hi16/2 .. 16+hi16/2+7]
static __device__ __forceinline__ v16h bfrag_f16(const f16* p, int hi16) {
  v8h a = *(const v8h*)(p + (hi16 >> 1));
  v8h b = *(const v8h*)(p + 16 + (hi16 >> 1));
  v16h r;
#pragma unroll
  for (int i = 0; i < 8; ++i) { r[i] = a[i]; r[i + 8] = b[i]; }
  return r;
}

// ============================================================================
// Kernel 0: one-shot f32 -> f16 conversion (x, qkv_w, proj_w).
// Each thread converts 8 elements: 2x b128 loads -> 1x b128 store.
// ============================================================================
__global__ __launch_bounds__(256) void k_cvt(const float* __restrict__ in,
                                             f16* __restrict__ out, int n8)
{
  int idx = blockIdx.x * 256 + threadIdx.x;
  if (idx >= n8) return;
  const float* p = in + (size_t)idx * 8;
  v4f a = *(const v4f*)p;
  v4f b = *(const v4f*)(p + 4);
  v8h o;
#pragma unroll
  for (int i = 0; i < 4; ++i) { o[i] = (f16)a[i]; o[i + 4] = (f16)b[i]; }
  *(v8h*)(out + (size_t)idx * 8) = o;
}

// ============================================================================
// Kernel 1: QKV projection.  Y[8192,3072] = Xh[8192,1024] @ Wq^T + b  (f16 in)
// Scatter into Qh,Kh [BH,T,D] (f16) and Vt [BH,D,T] (f16, transposed).
// Wave tile 32x64 (2x4 WMMA C-frags); block 256 threads = 8 waves -> 64x256.
// ============================================================================
__global__ __launch_bounds__(256) void k_qkv(
    const f16* __restrict__ x, const f16* __restrict__ w,
    const float* __restrict__ bias,
    f16* __restrict__ Qh, f16* __restrict__ Kh, f16* __restrict__ Vt)
{
  const int lane = threadIdx.x & 31;
  const int wav  = threadIdx.x >> 5;
  const int l15  = lane & 15;
  const int hi   = lane >> 4;
  const int kb8  = hi * 8;
  const int hi16 = hi * 16;
  const int m0 = blockIdx.y * 64  + (wav >> 2) * 32;
  const int n0 = blockIdx.x * 256 + (wav & 3) * 64;

  const f16* xr[2];
  const f16* wr[4];
#pragma unroll
  for (int mt = 0; mt < 2; ++mt) xr[mt] = x + (size_t)(m0 + mt * 16 + l15) * C_;
#pragma unroll
  for (int nt = 0; nt < 4; ++nt) wr[nt] = w + (size_t)(n0 + nt * 16 + l15) * C_;

  v8f acc[2][4];
#pragma unroll
  for (int i = 0; i < 2; ++i)
#pragma unroll
    for (int j = 0; j < 4; ++j) acc[i][j] = zero8();

  for (int kk = 0; kk < C_; kk += 32) {
    if (kk + 256 < C_) {           // wave-uniform; gfx1250 global_prefetch_b8
#pragma unroll
      for (int mt = 0; mt < 2; ++mt) __builtin_prefetch(xr[mt] + kk + 256, 0, 1);
#pragma unroll
      for (int nt = 0; nt < 4; ++nt) __builtin_prefetch(wr[nt] + kk + 256, 0, 1);
    }
    v16h a[2];
#pragma unroll
    for (int mt = 0; mt < 2; ++mt) a[mt] = afrag_f16(xr[mt] + kk, kb8);
    v16h b[4];
#pragma unroll
    for (int nt = 0; nt < 4; ++nt) b[nt] = bfrag_f16(wr[nt] + kk, hi16);
#pragma unroll
    for (int mt = 0; mt < 2; ++mt)
#pragma unroll
      for (int nt = 0; nt < 4; ++nt)
        acc[mt][nt] = wmma16(a[mt], b[nt], acc[mt][nt]);
  }

#pragma unroll
  for (int mt = 0; mt < 2; ++mt)
#pragma unroll
    for (int nt = 0; nt < 4; ++nt)
#pragma unroll
      for (int r = 0; r < 8; ++r) {
        int mg = m0 + mt * 16 + r + hi * 8;
        int ng = n0 + nt * 16 + l15;
        float v = acc[mt][nt][r] + bias[ng];
        int s = ng >> 10, c = ng & 1023, h = c >> 6, d = c & 63;
        int b = mg >> 11, t = mg & 2047;
        int bh = b * H_ + h;
        f16 hv = (f16)v;
        if (s == 0)      Qh[((size_t)(bh * T_ + t)) * D_ + d] = hv;
        else if (s == 1) Kh[((size_t)(bh * T_ + t)) * D_ + d] = hv;
        else             Vt[((size_t)(bh * D_ + d)) * T_ + t] = hv;
      }
}

// ============================================================================
// Kernel 2: windowed-causal flash attention. One wave = one 16-query tile.
// Block 128 threads = 4 independent wave-tiles. Online softmax over key
// tiles of 32; P transposed C-layout -> A-layout through per-wave LDS.
// ============================================================================
__global__ __launch_bounds__(128) void k_attn(
    const f16* __restrict__ Qh, const f16* __restrict__ Kh,
    const f16* __restrict__ Vt, f16* __restrict__ At)
{
  __shared__ f16 plds[4 * 16 * 32];
  const int lane = threadIdx.x & 31;
  const int wav  = threadIdx.x >> 5;
  const int l15  = lane & 15;
  const int hi   = lane >> 4;
  const int kb8  = hi * 8;
  const int hi16 = hi * 16;

  const int tile = blockIdx.x * 4 + wav;   // 0 .. B*H*(T/16)-1
  const int bh = tile >> 7;                // 128 q-tiles per (b,h)
  const int q0 = (tile & 127) * 16;

  const f16* Qp = Qh + (size_t)bh * T_ * D_;
  const f16* Kp = Kh + (size_t)bh * T_ * D_;
  const f16* Vp = Vt + (size_t)bh * D_ * T_;
  f16* pl = plds + wav * (16 * 32);

  v16h aq[2];
  aq[0] = afrag_f16(Qp + (size_t)(q0 + l15) * D_,      kb8);
  aq[1] = afrag_f16(Qp + (size_t)(q0 + l15) * D_ + 32, kb8);

  const float NEGINF = -__builtin_inff();
  float mrow[8], lrow[8];
#pragma unroll
  for (int r = 0; r < 8; ++r) { mrow[r] = NEGINF; lrow[r] = 0.f; }
  v8f o[4];
#pragma unroll
  for (int d = 0; d < 4; ++d) o[d] = zero8();

  int ks = q0 - (WIN_ - 1);
  if (ks < 0) ks = 0;
  for (int k0 = ks & ~31; k0 < q0 + 16; k0 += 32) {
    // prefetch next key/value tiles (addresses stay inside workspace)
    __builtin_prefetch(Kp + (size_t)(k0 + 32 + l15) * D_, 0, 1);
    __builtin_prefetch(Vp + (size_t)(hi * 32 + l15) * T_ + k0 + 32, 0, 1);

    // ---- scores S = Q @ K^T for 32 keys (two 16-key N-subtiles)
    v8f sc[2];
#pragma unroll
    for (int n = 0; n < 2; ++n) {
      const f16* kbase = Kp + (size_t)(k0 + n * 16 + l15) * D_;
      v16h b0 = bfrag_f16(kbase,      hi16);
      v16h b1 = bfrag_f16(kbase + 32, hi16);
      v8f c = zero8();
      c = wmma16(aq[0], b0, c);
      c = wmma16(aq[1], b1, c);
      sc[n] = c;
    }
    // ---- mask, online softmax update (branchless; EXEC stays full)
#pragma unroll
    for (int r = 0; r < 8; ++r) {
      int qi = q0 + r + hi * 8;
#pragma unroll
      for (int n = 0; n < 2; ++n) {
        int kj = k0 + n * 16 + l15;
        bool ok = (kj <= qi) && (qi - kj < WIN_);
        float s = sc[n][r] * 0.125f;        // 1/sqrt(64)
        sc[n][r] = ok ? s : NEGINF;
      }
      float tm = fmaxf(sc[0][r], sc[1][r]);
#pragma unroll
      for (int m = 1; m < 16; m <<= 1) tm = fmaxf(tm, __shfl_xor(tm, m, 32));
      float mnew  = fmaxf(mrow[r], tm);
      float alpha = (mnew == NEGINF) ? 1.f : __expf(mrow[r] - mnew);
      float e0 = (sc[0][r] == NEGINF) ? 0.f : __expf(sc[0][r] - mnew);
      float e1 = (sc[1][r] == NEGINF) ? 0.f : __expf(sc[1][r] - mnew);
      float rs = e0 + e1;
#pragma unroll
      for (int m = 1; m < 16; m <<= 1) rs += __shfl_xor(rs, m, 32);
      lrow[r] = lrow[r] * alpha + rs;
      mrow[r] = mnew;
      sc[0][r] = e0; sc[1][r] = e1;
#pragma unroll
      for (int d = 0; d < 4; ++d) o[d][r] *= alpha;
    }
    // ---- P (C-layout) -> LDS 16x32 tile -> reload as A fragment
#pragma unroll
    for (int r = 0; r < 8; ++r) {
      pl[(r + hi * 8) * 32 + l15]      = (f16)sc[0][r];
      pl[(r + hi * 8) * 32 + 16 + l15] = (f16)sc[1][r];
    }
    asm volatile("s_wait_dscnt 0x0" ::: "memory");  // same-wave DS in-order; belt+braces
    v16h pf = afrag_f16(pl + l15 * 32, kb8);
    asm volatile("" ::: "memory");
    // ---- O += P @ V   (V transposed: contiguous over keys)
#pragma unroll
    for (int d = 0; d < 4; ++d) {
      v16h bv = bfrag_f16(Vp + (size_t)(d * 16 + l15) * T_ + k0, hi16);
      o[d] = wmma16(pf, bv, o[d]);
    }
  }

  // ---- normalize and store to attn buffer [B*T, C] (f16)
  const int b = bh >> 4, h = bh & 15;
#pragma unroll
  for (int r = 0; r < 8; ++r) {
    float inv = 1.f / lrow[r];
    int trow = b * T_ + q0 + r + hi * 8;
#pragma unroll
    for (int d = 0; d < 4; ++d) {
      int col = h * D_ + d * 16 + l15;
      At[(size_t)trow * C_ + col] = (f16)(o[d][r] * inv);
    }
  }
}

// ============================================================================
// Kernel 3: output projection. out[8192,1024] = At @ Wp^T + proj_b (f32 out)
// ============================================================================
__global__ __launch_bounds__(256) void k_proj(
    const f16* __restrict__ At, const f16* __restrict__ w,
    const float* __restrict__ bias, float* __restrict__ out)
{
  const int lane = threadIdx.x & 31;
  const int wav  = threadIdx.x >> 5;
  const int l15  = lane & 15;
  const int hi   = lane >> 4;
  const int kb8  = hi * 8;
  const int hi16 = hi * 16;
  const int m0 = blockIdx.y * 64  + (wav >> 2) * 32;
  const int n0 = blockIdx.x * 256 + (wav & 3) * 64;

  const f16* ar[2];
  const f16* wr[4];
#pragma unroll
  for (int mt = 0; mt < 2; ++mt) ar[mt] = At + (size_t)(m0 + mt * 16 + l15) * C_;
#pragma unroll
  for (int nt = 0; nt < 4; ++nt) wr[nt] = w + (size_t)(n0 + nt * 16 + l15) * C_;

  v8f acc[2][4];
#pragma unroll
  for (int i = 0; i < 2; ++i)
#pragma unroll
    for (int j = 0; j < 4; ++j) acc[i][j] = zero8();

  for (int kk = 0; kk < C_; kk += 32) {
    if (kk + 256 < C_) {
#pragma unroll
      for (int mt = 0; mt < 2; ++mt) __builtin_prefetch(ar[mt] + kk + 256, 0, 1);
#pragma unroll
      for (int nt = 0; nt < 4; ++nt) __builtin_prefetch(wr[nt] + kk + 256, 0, 1);
    }
    v16h a[2];
#pragma unroll
    for (int mt = 0; mt < 2; ++mt) a[mt] = afrag_f16(ar[mt] + kk, kb8);
    v16h b[4];
#pragma unroll
    for (int nt = 0; nt < 4; ++nt) b[nt] = bfrag_f16(wr[nt] + kk, hi16);
#pragma unroll
    for (int mt = 0; mt < 2; ++mt)
#pragma unroll
      for (int nt = 0; nt < 4; ++nt)
        acc[mt][nt] = wmma16(a[mt], b[nt], acc[mt][nt]);
  }

#pragma unroll
  for (int mt = 0; mt < 2; ++mt)
#pragma unroll
    for (int nt = 0; nt < 4; ++nt)
#pragma unroll
      for (int r = 0; r < 8; ++r) {
        int mg = m0 + mt * 16 + r + hi * 8;
        int ng = n0 + nt * 16 + l15;
        out[(size_t)mg * C_ + ng] = acc[mt][nt][r] + bias[ng];
      }
}

// ============================================================================
extern "C" void kernel_launch(void* const* d_in, const int* in_sizes, int n_in,
                              void* d_out, int out_size, void* d_ws, size_t ws_size,
                              hipStream_t stream) {
  (void)in_sizes; (void)n_in; (void)out_size; (void)ws_size;
  const float* x      = (const float*)d_in[0];
  // d_in[1] = attn_mask (bool) — mask is derived analytically from WINDOW
  const float* qkv_w  = (const float*)d_in[2];
  const float* qkv_b  = (const float*)d_in[3];
  const float* proj_w = (const float*)d_in[4];
  const float* proj_b = (const float*)d_in[5];
  float* out = (float*)d_out;

  const size_t SZ  = (size_t)B_ * H_ * T_ * D_;  // 8,388,608 halves (16 MB)
  const size_t XS  = (size_t)M_ * C_;            // 8,388,608
  const size_t WQS = (size_t)N3_ * C_;           // 3,145,728
  const size_t WPS = (size_t)C_ * C_;            // 1,048,576
  f16* Qh = (f16*)d_ws;
  f16* Kh = Qh + SZ;
  f16* Vt = Kh + SZ;
  f16* At = Vt + SZ;
  f16* Xh = At + SZ;
  f16* Wq = Xh + XS;
  f16* Wp = Wq + WQS;   // total workspace use: ~88 MB

  // f32 -> f16 staging (one pass each; weights/activations reused many times)
  k_cvt<<<dim3((XS  / 8 + 255) / 256), 256, 0, stream>>>(x,      Xh, (int)(XS  / 8));
  k_cvt<<<dim3((WQS / 8 + 255) / 256), 256, 0, stream>>>(qkv_w,  Wq, (int)(WQS / 8));
  k_cvt<<<dim3((WPS / 8 + 255) / 256), 256, 0, stream>>>(proj_w, Wp, (int)(WPS / 8));

  dim3 g1(N3_ / 256, M_ / 64);            // 12 x 128
  k_qkv<<<g1, 256, 0, stream>>>(Xh, Wq, qkv_b, Qh, Kh, Vt);

  dim3 g2(B_ * H_ * (T_ / 16) / 4);       // 2048 blocks x 4 waves
  k_attn<<<g2, 128, 0, stream>>>(Qh, Kh, Vt, At);

  dim3 g3(C_ / 256, M_ / 64);             // 4 x 128
  k_proj<<<g3, 256, 0, stream>>>(At, Wp, proj_b, out);
}
